// EncodeProcessDecodeMultiScale_27496380629766
// MI455X (gfx1250) — matrix operand order repaired and checked by reading, not compiled, not run on hardware
//
#include <hip/hip_runtime.h>
#include <hip/hip_bf16.h>
#include <hip/hip_fp16.h>

typedef __attribute__((ext_vector_type(16))) _Float16 v16h;
typedef __attribute__((ext_vector_type(8)))  _Float16 v8h;
typedef __attribute__((ext_vector_type(8)))  float    v8f;
typedef _Float16 h16;

#define HDIM 128
#define NN_  40000
#define EE_  240000
#define NC_  20000
#define EC_  60000

// ---------------- fused GEMM + bias + (ReLU) + (LayerNorm) + (residual) ----------------
// Activations live in global memory as f16; WMMA consumes f16 anyway, epilogue math is f32.
// A-operand modes:
//   0: plain f16 (din==128)   A[i][j] = a0[i*128 + j]
//   1: normalized f32 raw     A[i][j] = (araw[i*din+j]-mean[j])/std[j]   (din = 7 or 11)
//   2: edge concat din=384:   [ nf[row[i]] | nf[col[i]] | ef[i] ]   (all f16)
//   3: concat2   din=256:     [ a0[i] | a1[i] ]                     (f16)
// epimode: 0 = ReLU only; 1 = LN (no relu); 2 = ReLU then LN
// resid (f16): added AFTER LN. If out2 != null: out1 = val (pre-residual), out2 = val+resid;
//              else out1 = val (+resid if resid != null).
__global__ __launch_bounds__(256)
void gemm128(const h16* __restrict__ a0, const h16* __restrict__ a1,
             const h16* __restrict__ a2, const float* __restrict__ araw,
             const int* __restrict__ idx0, const int* __restrict__ idx1,
             const float* __restrict__ mean, const float* __restrict__ stdv,
             int M, int din, int amode,
             const float* __restrict__ W, const float* __restrict__ b,
             const float* __restrict__ g, const float* __restrict__ beta,
             const h16* __restrict__ resid,
             h16* __restrict__ out1, h16* __restrict__ out2,
             int epimode) {
    // double-buffered tiles, padded row stride (40 halves = 80B, 16B-aligned rows)
    __shared__ __attribute__((aligned(16))) _Float16 sA[2][128][40];
    __shared__ __attribute__((aligned(16))) _Float16 sW[2][128][40];
    __shared__ int sI0[128], sI1[128];

    const int tid  = threadIdx.x;
    const int wave = tid >> 5;
    const int lane = tid & 31;
    const long long rowBase = (long long)blockIdx.x * 128;
    const int waveRow = wave * 16;

    if (amode == 2) {   // stage gather indices once per block
        if (tid < 128) {
            long long gi = rowBase + tid;
            sI0[tid] = (gi < M) ? idx0[gi] : 0;
        } else {
            long long gi = rowBase + (tid - 128);
            sI1[tid - 128] = (gi < M) ? idx1[gi] : 0;
        }
    }
    __syncthreads();

    auto stage = [&](int kc, int buf) {
        if (amode == 1) {
            // f32 raw input, tiny din (7/11): scalar with fused normalization
            for (int t = tid; t < 1024; t += 256) {
                const int r  = t >> 3;
                const int k4 = (t & 7) * 4;
                const long long gi = rowBase + r;
#pragma unroll
                for (int i = 0; i < 4; ++i) {
                    const int j = kc * 32 + k4 + i;
                    float v = 0.0f;
                    if (gi < M && j < din)
                        v = (araw[gi * din + j] - mean[j]) / stdv[j];
                    sA[buf][r][k4 + i] = (h16)v;
                }
            }
        } else {
            // f16 sources: 8 halves per group = one b128 global load + one b128 LDS store
            for (int t = tid; t < 512; t += 256) {
                const int r  = t >> 2;
                const int k8 = (t & 3) * 8;
                const int j  = kc * 32 + k8;
                const long long gi = rowBase + r;
                v8h v = {};
                if (gi < M) {
                    const h16* src;
                    if (amode == 0) {
                        src = &a0[gi * 128 + j];
                    } else if (amode == 2) {
                        if (j < 128)      src = &a0[(long long)sI0[r] * 128 + j];
                        else if (j < 256) src = &a0[(long long)sI1[r] * 128 + (j - 128)];
                        else              src = &a2[gi * 128 + (j - 256)];
                    } else { // amode == 3
                        src = (j < 128) ? &a0[gi * 128 + j] : &a1[gi * 128 + (j - 128)];
                    }
                    v = *(const v8h*)src;
                }
                *(v8h*)&sA[buf][r][k8] = v;
            }
        }
        // W tile transposed: sW[n][k] = W[j][n] (f32 -> f16; weights tiny, L2-resident)
        for (int t = tid; t < 1024; t += 256) {
            const int jj = t >> 5;
            const int n4 = (t & 31) * 4;
            const int j  = kc * 32 + jj;
            float w0 = 0, w1 = 0, w2 = 0, w3 = 0;
            if (j < din) {
                const float* src = &W[(long long)j * 128 + n4];
                w0 = src[0]; w1 = src[1]; w2 = src[2]; w3 = src[3];
            }
            sW[buf][n4 + 0][jj] = (h16)w0;
            sW[buf][n4 + 1][jj] = (h16)w1;
            sW[buf][n4 + 2][jj] = (h16)w2;
            sW[buf][n4 + 3][jj] = (h16)w3;
        }
    };

    v8f acc[8];
#pragma unroll
    for (int i = 0; i < 8; ++i) acc[i] = (v8f)0.0f;

    const int nkc = (din + 31) / 32;
    stage(0, 0);
    int buf = 0;
    for (int kc = 0; kc < nkc; ++kc) {
        __syncthreads();
        if (kc + 1 < nkc) stage(kc + 1, buf ^ 1);   // overlap next tile's global loads with WMMA

        const int m   = waveRow + (lane & 15);
        const int akb = (lane < 16) ? 0 : 8;
        v16h afrag;
        ((v8h*)&afrag)[0] = *(const v8h*)&sA[buf][m][akb];
        ((v8h*)&afrag)[1] = *(const v8h*)&sA[buf][m][akb + 16];

        const int nn  = lane & 15;
        const int bkb = (lane < 16) ? 0 : 8;
#pragma unroll
        for (int nt = 0; nt < 8; ++nt) {
            v16h bfrag;
            ((v8h*)&bfrag)[0] = *(const v8h*)&sW[buf][nt * 16 + nn][bkb];
            ((v8h*)&bfrag)[1] = *(const v8h*)&sW[buf][nt * 16 + nn][bkb + 16];
            acc[nt] = __builtin_amdgcn_wmma_f32_16x16x32_f16(
                false, afrag, false, bfrag, (short)0, acc[nt], false, false);
        }
        buf ^= 1;
    }

    // -------- epilogue: bias, relu, LayerNorm (full 128-wide row lives in this block) --------
    const bool do_relu = (epimode == 0 || epimode == 2);
    const bool do_ln   = (epimode >= 1);
    const int  colBase = lane & 15;

    float bcol[8], gcol[8], betacol[8];
#pragma unroll
    for (int nt = 0; nt < 8; ++nt) {
        const int col = nt * 16 + colBase;
        bcol[nt]    = b[col];
        gcol[nt]    = do_ln ? g[col]    : 0.0f;
        betacol[nt] = do_ln ? beta[col] : 0.0f;
    }

#pragma unroll
    for (int v = 0; v < 8; ++v) {
        const int r = waveRow + v + ((lane < 16) ? 0 : 8);
        const long long gRow = rowBase + r;
        float t[8], sum = 0.0f, sq = 0.0f;
#pragma unroll
        for (int nt = 0; nt < 8; ++nt) {
            float x = acc[nt][v] + bcol[nt];
            if (do_relu) x = fmaxf(x, 0.0f);
            t[nt] = x;
            sum += x; sq += x * x;
        }
        float mu = 0.0f, rs = 1.0f;
        if (do_ln) {
#pragma unroll
            for (int msk = 1; msk <= 8; msk <<= 1) {   // row spans lanes 0-15 / 16-31
                sum += __shfl_xor(sum, msk, 32);
                sq  += __shfl_xor(sq,  msk, 32);
            }
            mu = sum * (1.0f / 128.0f);
            float var = sq * (1.0f / 128.0f) - mu * mu;
            rs = rsqrtf(var + 1e-5f);
        }
        if (gRow < M) {
#pragma unroll
            for (int nt = 0; nt < 8; ++nt) {
                const int col = nt * 16 + colBase;
                float val = t[nt];
                if (do_ln) val = (val - mu) * rs * gcol[nt] + betacol[nt];
                const long long o = gRow * 128 + col;
                if (out2) {
                    out1[o] = (h16)val;
                    out2[o] = (h16)(val + (float)resid[o]);
                } else if (resid) {
                    out1[o] = (h16)(val + (float)resid[o]);
                } else {
                    out1[o] = (h16)val;
                }
            }
        }
    }
}

// ---------------- helper kernels (f16 buffers handled as 32-bit words) ----------------
__global__ void zero_u32(unsigned int* p, long long n) {
    long long i = (long long)blockIdx.x * blockDim.x + threadIdx.x;
    if (i < n) p[i] = 0u;
}

// rows are 128 halves = 64 dwords
__global__ void gather_rows(const unsigned int* __restrict__ src, const int* __restrict__ fi,
                            unsigned int* __restrict__ dst, int nrows) {
    long long i = (long long)blockIdx.x * blockDim.x + threadIdx.x;
    if (i < (long long)nrows * 64) {
        int j = (int)(i >> 6), c = (int)(i & 63);
        dst[i] = src[(long long)fi[j] * 64 + c];
    }
}

__global__ void scatter_rows(const unsigned int* __restrict__ src, const int* __restrict__ fi,
                             unsigned int* __restrict__ dst, int nrows) {
    long long i = (long long)blockIdx.x * blockDim.x + threadIdx.x;
    if (i < (long long)nrows * 64) {
        int j = (int)(i >> 6), c = (int)(i & 63);
        dst[(long long)fi[j] * 64 + c] = src[i];
    }
}

// aggr[col[e]] += newe[e]  via packed-f16 atomics (global_atomic_pk_add_f16)
__global__ void scatter_add(const __half2* __restrict__ newe, const int* __restrict__ col,
                            __half2* __restrict__ aggr, int nedges) {
    long long i = (long long)blockIdx.x * blockDim.x + threadIdx.x;
    if (i < (long long)nedges * 64) {
        int e = (int)(i >> 6), c = (int)(i & 63);
        atomicAdd(&aggr[(long long)col[e] * 64 + c], newe[i]);
    }
}

// decoder tail: out[r][o] = b2[o] + sum_c h[r][c] * w2[c][o], OUT=3 (T=1, dt=1)
__global__ void dec_tail(const h16* __restrict__ h, const float* __restrict__ w2,
                         const float* __restrict__ b2, float* __restrict__ out, int M) {
    long long i = (long long)blockIdx.x * blockDim.x + threadIdx.x;
    if (i < (long long)M * 3) {
        int r = (int)(i / 3), o = (int)(i % 3);
        float s = b2[o];
        const h16* hr = h + (long long)r * 128;
#pragma unroll 8
        for (int c = 0; c < 128; ++c) s += (float)hr[c] * w2[c * 3 + o];
        out[i] = s;
    }
}

// ---------------- host side ----------------
struct MLPw { const float *w1, *b1, *w2, *b2, *g, *beta; };

extern "C" void kernel_launch(void* const* d_in, const int* in_sizes, int n_in,
                              void* d_out, int out_size, void* d_ws, size_t ws_size,
                              hipStream_t stream) {
    (void)in_sizes; (void)out_size; (void)ws_size;
    const float* x  = (const float*)d_in[0];
    const float* e  = (const float*)d_in[1];
    const float* ce = (const float*)d_in[2];
    const int* edge_index   = (const int*)d_in[3];
    const int* c_edge_index = (const int*)d_in[4];
    const int* first_idx    = (const int*)d_in[5];

    const float* P[128];
    for (int i = 6; i < n_in && i < 6 + 100; ++i) P[i] = (const float*)d_in[i];
    // params flattened (jax tree: dict keys sorted, MLP keys sorted b1,b2,beta,g,w1,w2):
    auto mlp6 = [&](int base) { MLPw m; m.b1 = P[base]; m.b2 = P[base+1]; m.beta = P[base+2];
                                m.g = P[base+3]; m.w1 = P[base+4]; m.w2 = P[base+5]; return m; };
    MLPw cedge_enc = mlp6(6);
    const float* cedge_mean = P[12]; const float* cedge_std = P[13];
    MLPw coarse_e[3], coarse_n[3], fine_e[3], fine_n[3];
    for (int s = 0; s < 3; ++s) { coarse_e[s] = mlp6(14 + s*12); coarse_n[s] = mlp6(20 + s*12); }
    const float* dec_b1 = P[50]; const float* dec_b2 = P[51];
    const float* dec_w1 = P[52]; const float* dec_w2 = P[53];
    MLPw edge_enc = mlp6(54);
    const float* edge_mean = P[60]; const float* edge_std = P[61];
    for (int s = 0; s < 3; ++s) { fine_e[s] = mlp6(62 + s*12); fine_n[s] = mlp6(68 + s*12); }
    MLPw node_enc = mlp6(98);
    const float* node_mean = P[104]; const float* node_std = P[105];

    const int* e_row = edge_index;   const int* e_col = edge_index + EE_;
    const int* c_row = c_edge_index; const int* c_col = c_edge_index + EC_;

    // workspace layout: all activations f16 (halves the dominant HBM traffic)
    h16* w = (h16*)d_ws;
    long long off = 0;
    auto take = [&](long long rows) { h16* p = w + off; off += rows * 128; return p; };
    h16* xhA  = take(NN_); h16* xhB  = take(NN_);
    h16* ehA  = take(EE_); h16* ehB  = take(EE_);
    h16* cehA = take(EC_); h16* cehB = take(EC_);
    h16* cxhA = take(NC_); h16* cxhB = take(NC_);
    h16* hidden = take(EE_);
    h16* newe   = take(EE_);
    h16* aggr   = take(NN_);
    h16* mapb   = take(NN_);

    auto gemm = [&](const h16* a0, const h16* a1, const h16* a2, const float* araw,
                    const int* i0, const int* i1,
                    const float* mean, const float* stdv,
                    int M, int din, int amode,
                    const MLPw* mw, int layer, const h16* resid,
                    h16* o1, h16* o2, int epi) {
        const float* W = (layer == 1) ? mw->w1 : mw->w2;
        const float* b = (layer == 1) ? mw->b1 : mw->b2;
        dim3 grid((M + 127) / 128);
        gemm128<<<grid, 256, 0, stream>>>(a0, a1, a2, araw, i0, i1, mean, stdv, M, din, amode,
                                          W, b, mw->g, mw->beta, resid, o1, o2, epi);
    };
    auto zero = [&](h16* p, long long nhalves) {
        long long nw = nhalves / 2;
        zero_u32<<<dim3((unsigned)((nw + 255) / 256)), 256, 0, stream>>>((unsigned int*)p, nw);
    };

    // ---- encoders ----
    gemm(nullptr, nullptr, nullptr, x, nullptr, nullptr, node_mean, node_std,
         NN_, 11, 1, &node_enc, 1, nullptr, hidden, nullptr, 0);
    gemm(hidden, nullptr, nullptr, nullptr, nullptr, nullptr, nullptr, nullptr,
         NN_, 128, 0, &node_enc, 2, nullptr, xhA, nullptr, 1);
    gemm(nullptr, nullptr, nullptr, e, nullptr, nullptr, edge_mean, edge_std,
         EE_, 7, 1, &edge_enc, 1, nullptr, hidden, nullptr, 0);
    gemm(hidden, nullptr, nullptr, nullptr, nullptr, nullptr, nullptr, nullptr,
         EE_, 128, 0, &edge_enc, 2, nullptr, ehA, nullptr, 1);
    gemm(nullptr, nullptr, nullptr, ce, nullptr, nullptr, cedge_mean, cedge_std,
         EC_, 7, 1, &cedge_enc, 1, nullptr, hidden, nullptr, 0);
    gemm(hidden, nullptr, nullptr, nullptr, nullptr, nullptr, nullptr, nullptr,
         EC_, 128, 0, &cedge_enc, 2, nullptr, cehA, nullptr, 1);

    // cx_h = x_h[first_indices]
    gather_rows<<<dim3((NC_ * 64 + 255) / 256), 256, 0, stream>>>(
        (const unsigned int*)xhA, first_idx, (unsigned int*)cxhA, NC_);

    // ---- fine processor: 3 steps ----
    h16* xc = xhA; h16* xn = xhB; h16* ec = ehA; h16* en = ehB;
    for (int s = 0; s < 3; ++s) {
        gemm(xc, nullptr, ec, nullptr, e_row, e_col, nullptr, nullptr,
             EE_, 384, 2, &fine_e[s], 1, nullptr, hidden, nullptr, 0);
        gemm(hidden, nullptr, nullptr, nullptr, nullptr, nullptr, nullptr, nullptr,
             EE_, 128, 0, &fine_e[s], 2, ec, newe, en, 2);
        zero(aggr, (long long)NN_ * 128);
        scatter_add<<<dim3(((long long)EE_ * 64 + 255) / 256), 256, 0, stream>>>(
            (const __half2*)newe, e_col, (__half2*)aggr, EE_);
        gemm(xc, aggr, nullptr, nullptr, nullptr, nullptr, nullptr, nullptr,
             NN_, 256, 3, &fine_n[s], 1, nullptr, hidden, nullptr, 0);
        gemm(hidden, nullptr, nullptr, nullptr, nullptr, nullptr, nullptr, nullptr,
             NN_, 128, 0, &fine_n[s], 2, xc, xn, nullptr, 2);
        h16* t = xc; xc = xn; xn = t;  t = ec; ec = en; en = t;
    }

    // ---- coarse processor: 3 steps ----
    h16* cxc = cxhA; h16* cxn = cxhB; h16* cec = cehA; h16* cen = cehB;
    for (int s = 0; s < 3; ++s) {
        gemm(cxc, nullptr, cec, nullptr, c_row, c_col, nullptr, nullptr,
             EC_, 384, 2, &coarse_e[s], 1, nullptr, hidden, nullptr, 0);
        gemm(hidden, nullptr, nullptr, nullptr, nullptr, nullptr, nullptr, nullptr,
             EC_, 128, 0, &coarse_e[s], 2, cec, newe, cen, 2);
        zero(aggr, (long long)NC_ * 128);
        scatter_add<<<dim3(((long long)EC_ * 64 + 255) / 256), 256, 0, stream>>>(
            (const __half2*)newe, c_col, (__half2*)aggr, EC_);
        gemm(cxc, aggr, nullptr, nullptr, nullptr, nullptr, nullptr, nullptr,
             NC_, 256, 3, &coarse_n[s], 1, nullptr, hidden, nullptr, 0);
        gemm(hidden, nullptr, nullptr, nullptr, nullptr, nullptr, nullptr, nullptr,
             NC_, 128, 0, &coarse_n[s], 2, cxc, cxn, nullptr, 2);
        h16* t = cxc; cxc = cxn; cxn = t;  t = cec; cec = cen; cen = t;
    }

    // ---- map back + decoder ----
    zero(mapb, (long long)NN_ * 128);
    scatter_rows<<<dim3((NC_ * 64 + 255) / 256), 256, 0, stream>>>(
        (const unsigned int*)cxc, first_idx, (unsigned int*)mapb, NC_);

    MLPw dec; dec.w1 = dec_w1; dec.b1 = dec_b1; dec.w2 = dec_w2; dec.b2 = dec_b2;
    dec.g = nullptr; dec.beta = nullptr;
    gemm(xc, mapb, nullptr, nullptr, nullptr, nullptr, nullptr, nullptr,
         NN_, 256, 3, &dec, 1, nullptr, hidden, nullptr, 0);
    dec_tail<<<dim3((NN_ * 3 + 255) / 256), 256, 0, stream>>>(hidden, dec_w2, dec_b2,
                                                              (float*)d_out, NN_);
}
